// MultiHeadAttention_38371237823145
// MI455X (gfx1250) — compile-verified, matrix operand order checked by reading, not run on hardware
//
#include <hip/hip_runtime.h>
#include <hip/hip_bf16.h>

#define S_LEN 2048
#define DE    1024
#define NH    16
#define DH    64
#define NB    4

typedef __attribute__((ext_vector_type(16))) _Float16 v16h;
typedef __attribute__((ext_vector_type(8)))  _Float16 v8h;
typedef __attribute__((ext_vector_type(8)))  float    v8f;
typedef __attribute__((ext_vector_type(4)))  float    v4f;

// ---- WMMA helpers ------------------------------------------------------
__device__ __forceinline__ v8f wmma_f16(v16h a, v16h b, v8f c) {
  // D = A*B + C, f32 accumulate. 8-arg form (neg_a, A, neg_b, B, c_mod, C, reuse_a, reuse_b)
  return __builtin_amdgcn_wmma_f32_16x16x32_f16(false, a, false, b, (short)0, c, false, false);
}

// Load a 16x32 f16 A/B fragment from row-major memory whose contiguous dim is K.
// Caller passes the lane's row pointer; kb = k0 + (lane>>4)*8.
// Lane layout (ISA 7.12.2): elems 0..7 = K[kb..kb+7], elems 8..15 = K[kb+16..kb+23].
__device__ __forceinline__ v16h load_frag_h(const _Float16* rowp, int kb) {
  v8h lo = *(const v8h*)(rowp + kb);
  v8h hi = *(const v8h*)(rowp + kb + 16);
  return __builtin_shufflevector(lo, hi, 0,1,2,3,4,5,6,7,8,9,10,11,12,13,14,15);
}

// Same fragment, but source data is f32 (global or LDS); convert to f16 in-register.
__device__ __forceinline__ v16h load_frag_f32(const float* rowp, int kb) {
  v4f a0 = *(const v4f*)(rowp + kb);
  v4f a1 = *(const v4f*)(rowp + kb + 4);
  v4f b0 = *(const v4f*)(rowp + kb + 16);
  v4f b1 = *(const v4f*)(rowp + kb + 20);
  v16h r;
#pragma unroll
  for (int i = 0; i < 4; ++i) {
    r[i]      = (_Float16)a0[i];
    r[4 + i]  = (_Float16)a1[i];
    r[8 + i]  = (_Float16)b0[i];
    r[12 + i] = (_Float16)b1[i];
  }
  return r;
}

// ---- Kernel 0: weights f32 [k][c] -> f16 transposed [c][k] -------------
// f32 weights are read exactly once: non-temporal load keeps them out of L2;
// the f16 transposed copies ARE reused heavily and stay RT.
__global__ __launch_bounds__(256) void wt_kernel(const float* __restrict__ W0,
                                                 const float* __restrict__ W1,
                                                 const float* __restrict__ W2,
                                                 const float* __restrict__ W3,
                                                 _Float16* __restrict__ Wt) {
  const float* Ws[4] = {W0, W1, W2, W3};
  int wsel = blockIdx.y;
  const float* W = Ws[wsel];
  _Float16* dst = Wt + (size_t)wsel * DE * DE;
  int idx = blockIdx.x * 256 + threadIdx.x;   // coalesced read of W
  int k = idx >> 10, c = idx & 1023;
  float wv = __builtin_nontemporal_load(&W[idx]);
  dst[(size_t)c * DE + k] = (_Float16)wv;
}

// ---- Kernel 1: QKV projections (grid.z selects Q/K/V) ------------------
// X[8192,1024]f32 @ Wt^T -> q_ws/k_ws [b,h,s,64]f16, v transposed -> vT [b,h,64,s]f16
__global__ __launch_bounds__(128) void proj_kernel(
    const float* __restrict__ Q, const float* __restrict__ K, const float* __restrict__ V,
    const _Float16* __restrict__ Wt,
    const float* __restrict__ bq, const float* __restrict__ bk, const float* __restrict__ bv,
    _Float16* __restrict__ q_ws, _Float16* __restrict__ k_ws, _Float16* __restrict__ vT_ws) {
  const int which = blockIdx.z;
  const float* X = (which == 0) ? Q : (which == 1) ? K : V;
  const _Float16* Wtw = Wt + (size_t)which * DE * DE;
  const float* bias = (which == 0) ? bq : (which == 1) ? bk : bv;

  const int tid = threadIdx.x, lane = tid & 31, w = tid >> 5;
  const int r0 = blockIdx.y * 64 + w * 16;
  const int c0 = blockIdx.x * 64;
  const int rlane = lane & 15;
  const int kb8 = (lane >> 4) * 8;

  v8f acc[4] = {};
  const float* arow = X + (size_t)(r0 + rlane) * DE;
  for (int k = 0; k < DE; k += 32) {
    v16h a = load_frag_f32(arow, k + kb8);
#pragma unroll
    for (int j = 0; j < 4; ++j) {
      const _Float16* brow = Wtw + (size_t)(c0 + j * 16 + rlane) * DE;
      v16h b = load_frag_h(brow, k + kb8);
      acc[j] = wmma_f16(a, b, acc[j]);
    }
  }
#pragma unroll
  for (int j = 0; j < 4; ++j) {
    int c = c0 + j * 16 + rlane;
    float bval = bias[c];
    int h = c >> 6, d = c & 63;
#pragma unroll
    for (int v = 0; v < 8; ++v) {
      int r = r0 + v + 8 * (lane >> 4);
      int bb = r >> 11, s = r & (S_LEN - 1);
      float val = acc[j][v] + bval;
      size_t bh = (size_t)bb * NH + h;
      if (which == 2)
        vT_ws[(bh * DH + d) * S_LEN + s] = (_Float16)val;
      else {
        _Float16* dst = (which == 0) ? q_ws : k_ws;
        dst[(bh * S_LEN + s) * DH + d] = (_Float16)val;
      }
    }
  }
}

// ---- Kernel 2: fused scores -> softmax -> attn write -> context --------
// One block = one (b,h) x 16 query rows; 4 waves; 16x2048 f32 score tile in LDS.
__global__ __launch_bounds__(128) void attn_kernel(
    const _Float16* __restrict__ q_ws, const _Float16* __restrict__ k_ws,
    const _Float16* __restrict__ vT_ws, const unsigned char* __restrict__ mask,
    float* __restrict__ attn_out, _Float16* __restrict__ ctx_ws) {
  __shared__ __align__(16) float sc[16][S_LEN + 4];   // +4 pad: kills LDS bank conflicts
  __shared__ float red[16][8];
  __shared__ float inv[16];

  const int tid = threadIdx.x, lane = tid & 31, w = tid >> 5;
  const int rlane = lane & 15;
  const int kb8 = (lane >> 4) * 8;
  const int qr0 = blockIdx.x * 16;
  const int bh = blockIdx.y;
  const int b = bh >> 4, h = bh & 15;

  // ---- Phase 1: scores = (q . k^T)/8 with mask, into LDS ----
  const _Float16* qrow = q_ws + ((size_t)bh * S_LEN + qr0 + rlane) * DH;
  v16h a0 = load_frag_h(qrow, kb8);
  v16h a1 = load_frag_h(qrow, 32 + kb8);
  const _Float16* kbase = k_ws + (size_t)bh * S_LEN * DH;
  const unsigned char* mbase = mask + (size_t)b * S_LEN * S_LEN;

  for (int t = 0; t < 32; ++t) {
    int n0 = w * 512 + t * 16;
    const _Float16* krow = kbase + (size_t)(n0 + rlane) * DH;
    v16h b0 = load_frag_h(krow, kb8);
    v16h b1 = load_frag_h(krow, 32 + kb8);
    v8f c = {};
    c = wmma_f16(a0, b0, c);
    c = wmma_f16(a1, b1, c);
#pragma unroll
    for (int v = 0; v < 8; ++v) {
      int m = v + 8 * (lane >> 4);
      int n = n0 + rlane;
      float sv = c[v] * 0.125f;                      // 1/sqrt(64)
      if (mbase[(size_t)(qr0 + m) * S_LEN + n]) sv = -1e9f;
      sc[m][n] = sv;
    }
  }
  __syncthreads();

  // ---- Phase 2: row softmax (8 threads per row) ----
  {
    int row = tid >> 3, j = tid & 7;
    float lm = -3.0e38f;
    for (int cc = j * 256; cc < j * 256 + 256; ++cc) lm = fmaxf(lm, sc[row][cc]);
    red[row][j] = lm;
    __syncthreads();
    float rm = red[row][0];
#pragma unroll
    for (int i = 1; i < 8; ++i) rm = fmaxf(rm, red[row][i]);
    float ls = 0.f;
    for (int cc = j * 256; cc < j * 256 + 256; ++cc) {
      float p = __expf(sc[row][cc] - rm);
      sc[row][cc] = p;
      ls += p;
    }
    __syncthreads();
    red[row][j] = ls;
    __syncthreads();
    if (j == 0) {
      float rs = 0.f;
#pragma unroll
      for (int i = 0; i < 8; ++i) rs += red[row][i];
      inv[row] = 1.0f / rs;
    }
    __syncthreads();
  }

  // ---- normalize in LDS + single coalesced NT write of attn to HBM ----
  // attn (1.07 GB) is never re-read: stream past L2 (TH=NT) so weights /
  // k/v panels / X stay L2-resident.
  float* aout = attn_out + (size_t)bh * S_LEN * S_LEN + (size_t)qr0 * S_LEN;
  for (int idx = tid; idx < 16 * S_LEN; idx += 128) {
    int row = idx >> 11, cc = idx & (S_LEN - 1);
    float p = sc[row][cc] * inv[row];
    sc[row][cc] = p;
    __builtin_nontemporal_store(p, &aout[(size_t)row * S_LEN + cc]);
  }
  __syncthreads();

  // ---- Phase 3: context = attn @ v (A from LDS f32->f16, B from vT) ----
  {
    int n0 = w * 16;
    const _Float16* vrow = vT_ws + ((size_t)bh * DH + n0 + rlane) * S_LEN;
    const float* srow = &sc[rlane][0];
    v8f acc = {};
    for (int kk = 0; kk < S_LEN; kk += 32) {
      v16h a = load_frag_f32(srow, kk + kb8);
      v16h bfr = load_frag_h(vrow, kk + kb8);
      acc = wmma_f16(a, bfr, acc);
    }
#pragma unroll
    for (int v = 0; v < 8; ++v) {
      int m = v + 8 * (lane >> 4);
      ctx_ws[((size_t)b * S_LEN + qr0 + m) * DE + h * DH + n0 + rlane] = (_Float16)acc[v];
    }
  }
}

// ---- Kernel 3: output projection out = ctx @ W_o + b_o (f32 out) -------
__global__ __launch_bounds__(128) void oproj_kernel(
    const _Float16* __restrict__ ctx_ws, const _Float16* __restrict__ WtO,
    const float* __restrict__ bo, float* __restrict__ out) {
  const int tid = threadIdx.x, lane = tid & 31, w = tid >> 5;
  const int r0 = blockIdx.y * 64 + w * 16;
  const int c0 = blockIdx.x * 64;
  const int rlane = lane & 15;
  const int kb8 = (lane >> 4) * 8;

  v8f acc[4] = {};
  const _Float16* arow = ctx_ws + (size_t)(r0 + rlane) * DE;
  for (int k = 0; k < DE; k += 32) {
    v16h a = load_frag_h(arow, k + kb8);
#pragma unroll
    for (int j = 0; j < 4; ++j) {
      const _Float16* brow = WtO + (size_t)(c0 + j * 16 + rlane) * DE;
      v16h bfr = load_frag_h(brow, k + kb8);
      acc[j] = wmma_f16(a, bfr, acc[j]);
    }
  }
#pragma unroll
  for (int j = 0; j < 4; ++j) {
    int c = c0 + j * 16 + rlane;
    float bval = bo[c];
#pragma unroll
    for (int v = 0; v < 8; ++v) {
      int r = r0 + v + 8 * (lane >> 4);
      // final output, no consumer on-device: stream past L2
      __builtin_nontemporal_store(acc[j][v] + bval, &out[(size_t)r * DE + c]);
    }
  }
}

extern "C" void kernel_launch(void* const* d_in, const int* in_sizes, int n_in,
                              void* d_out, int out_size, void* d_ws, size_t ws_size,
                              hipStream_t stream) {
  const float* Q = (const float*)d_in[0];
  const float* K = (const float*)d_in[1];
  const float* V = (const float*)d_in[2];
  const unsigned char* mask = (const unsigned char*)d_in[3];
  const float* W_q = (const float*)d_in[4];
  const float* b_q = (const float*)d_in[5];
  const float* W_k = (const float*)d_in[6];
  const float* b_k = (const float*)d_in[7];
  const float* W_v = (const float*)d_in[8];
  const float* b_v = (const float*)d_in[9];
  const float* W_o = (const float*)d_in[10];
  const float* b_o = (const float*)d_in[11];

  // workspace layout (f16): [Wq^T|Wk^T|Wv^T|Wo^T][q][k][vT][ctx]  (~72 MB)
  _Float16* wt = (_Float16*)d_ws;
  const size_t WT_ELEMS = 4ull * DE * DE;
  const size_t PROJ = (size_t)NB * NH * S_LEN * DH;   // 8,388,608
  _Float16* q_ws  = wt + WT_ELEMS;
  _Float16* k_ws  = q_ws + PROJ;
  _Float16* vT_ws = k_ws + PROJ;
  _Float16* ctx   = vT_ws + PROJ;

  float* out  = (float*)d_out;                         // [B,S,DE]
  float* attn = out + (size_t)NB * S_LEN * DE;         // [B,H,S,S]

  wt_kernel<<<dim3((DE * DE) / 256, 4), 256, 0, stream>>>(W_q, W_k, W_v, W_o, wt);
  proj_kernel<<<dim3(DE / 64, (NB * S_LEN) / 64, 3), 128, 0, stream>>>(
      Q, K, V, wt, b_q, b_k, b_v, q_ws, k_ws, vT_ws);
  attn_kernel<<<dim3(S_LEN / 16, NB * NH), 128, 0, stream>>>(
      q_ws, k_ws, vT_ws, mask, attn, ctx);
  oproj_kernel<<<dim3(DE / 64, (NB * S_LEN) / 64), 128, 0, stream>>>(
      ctx, wt + 3ull * DE * DE, b_o, out);
}
